// VanillaRNN_71236327571806
// MI455X (gfx1250) — compile-verified, matrix operand order checked by reading, not run on hardware
//
#include <hip/hip_runtime.h>
#include <math.h>

// Problem dims (fixed by the reference)
#define B_ 64
#define S_ 2048
#define I_ 128
#define H_ 256
#define O_ 128

typedef __bf16 bf16_t;
typedef __attribute__((ext_vector_type(16))) __bf16 v16bf;
typedef __attribute__((ext_vector_type(8)))  float  v8f;

// gfx1250 has a hardware V_TANH_F32 (TRANS32 op); use it if the toolchain
// exposes the builtin, else fall back to libm.
__device__ __forceinline__ float fast_tanh(float x) {
#if __has_builtin(__builtin_amdgcn_tanhf)
    return __builtin_amdgcn_tanhf(x);
#elif __has_builtin(__builtin_amdgcn_tanh_f32)
    return __builtin_amdgcn_tanh_f32(x);
#else
    return tanhf(x);
#endif
}

// ---------------------------------------------------------------------------
// Kernel 0: convert fp32 weights -> bf16 (row-major [n][k], which is exactly
// the contiguous-k layout this code's B-operand loads use).
// ---------------------------------------------------------------------------
__global__ void cvt_weights_kernel(const float* __restrict__ Wih,
                                   const float* __restrict__ Whh,
                                   const float* __restrict__ Wfc,
                                   bf16_t* __restrict__ wih,
                                   bf16_t* __restrict__ whh,
                                   bf16_t* __restrict__ wfc) {
    int i = blockIdx.x * blockDim.x + threadIdx.x;   // grid covers 65536
    if (i < H_ * I_) wih[i] = (bf16_t)Wih[i];
    if (i < H_ * H_) whh[i] = (bf16_t)Whh[i];
    if (i < O_ * H_) wfc[i] = (bf16_t)Wfc[i];
}

// ---------------------------------------------------------------------------
// Kernel 1: xproj[m,n] = sum_k x[m,k]*W_ih[n,k] + (b_ih[n]+b_hh[n])
//   M = B*S = 131072, K = I = 128, N = H = 256.
//   One 16x16 tile per wave, 4 k-steps of v_wmma_f32_16x16x32_bf16.
// ---------------------------------------------------------------------------
__global__ void __launch_bounds__(256)
xproj_kernel(const float* __restrict__ x, const bf16_t* __restrict__ wih,
             const float* __restrict__ b_ih, const float* __restrict__ b_hh,
             float* __restrict__ xproj) {
    const int lane  = threadIdx.x & 31;
    const int wid   = threadIdx.x >> 5;
    const int w     = blockIdx.x * 8 + wid;     // global tile id
    const int mtile = w >> 4;                   // 16 N-tiles
    const int ntile = w & 15;
    const int am    = mtile * 16 + (lane & 15); // A row for this lane
    const int n     = ntile * 16 + (lane & 15); // B column for this lane
    const int khalf = (lane >> 4) * 16;         // contiguous-k half per lane

    v8f acc = {};
#pragma unroll
    for (int kb = 0; kb < I_ / 32; ++kb) {
        const int k0 = kb * 32 + khalf;
        v16bf a;
        const float* xp = x + (size_t)am * I_ + k0;
#pragma unroll
        for (int j = 0; j < 16; ++j) a[j] = (bf16_t)xp[j];
        v16bf b = *(const v16bf*)(wih + (size_t)n * I_ + k0);
        acc = __builtin_amdgcn_wmma_f32_16x16x32_bf16(false, a, false, b,
                                                      (short)0, acc, false, false);
    }
    const float bias = b_ih[n] + b_hh[n];
    const int cmbase = mtile * 16 + ((lane >> 4) << 3);  // C rows: m = base + r
#pragma unroll
    for (int r = 0; r < 8; ++r)
        xproj[(size_t)(cmbase + r) * H_ + n] = acc[r] + bias;
}

// ---------------------------------------------------------------------------
// Kernel 2: the sequential scan.  4 workgroups (one batch-group of 16 rows
// each), 512 threads = 16 waves; wave w owns N-slice [16w,16w+16).
// W_hh is register-resident (64 VGPRs of B-operands per wave); h ping-pongs
// in LDS as bf16; one barrier per timestep.  xproj loads for step t+1 are
// issued before step t's WMMA chain so their latency hides under compute.
// ---------------------------------------------------------------------------
__global__ void __launch_bounds__(512)
rnn_scan_kernel(const float* __restrict__ xproj, const bf16_t* __restrict__ whh,
                float* __restrict__ hid) {
    __shared__ __align__(32) bf16_t hbuf[2][16][H_];   // 16 KB ping-pong

    const int lane  = threadIdx.x & 31;
    const int ntile = threadIdx.x >> 5;          // 0..15
    const int bg    = blockIdx.x;                // batch group 0..3
    const int khalf = (lane >> 4) * 16;
    const int n     = ntile * 16 + (lane & 15);  // this lane's output column
    const int am    = lane & 15;                 // A row within batch group
    const int cm    = (lane >> 4) << 3;          // C row base (0 or 8)

    // Hoist all of this wave's W_hh B-tiles into registers: Breg[kb] holds
    // the (32 x 16) tile for k in [32kb,32kb+32), column n; element (k,n) =
    // W_hh[n][k] -> contiguous 32B loads from row-major bf16 weights.
    v16bf Breg[H_ / 32];
#pragma unroll
    for (int kb = 0; kb < H_ / 32; ++kb)
        Breg[kb] = *(const v16bf*)(whh + (size_t)n * H_ + kb * 32 + khalf);

    // Per-lane base pointers into xproj / hidden for its 8 C rows; both
    // advance by H_ floats per timestep.
    const size_t brow = (size_t)bg * 16;
    const float* xpp[8];
    float*       hpp[8];
#pragma unroll
    for (int r = 0; r < 8; ++r) {
        const size_t row0 = (brow + cm + r) * (size_t)S_;   // t = 0
        xpp[r] = xproj + row0 * H_ + n;
        hpp[r] = hid   + row0 * H_ + n;
    }

    // h0 = 0
    for (int i = threadIdx.x; i < 16 * H_; i += 512)
        (&hbuf[0][0][0])[i] = (bf16_t)0.0f;

    // Prime the xproj pipeline for t = 0.
    float xp[8];
#pragma unroll
    for (int r = 0; r < 8; ++r) xp[r] = xpp[r][0];

    __syncthreads();

    int p = 0;
    for (int t = 0; t < S_; ++t) {
        // Kick off next step's xproj loads first (uniform clamped index:
        // no divergence, loads overlap the WMMA chain below).
        const int tn = (t + 1 < S_) ? (t + 1) : t;
        float xpn[8];
#pragma unroll
        for (int r = 0; r < 8; ++r) xpn[r] = xpp[r][(size_t)tn * H_];

        v8f acc = {};
#pragma unroll
        for (int kb = 0; kb < H_ / 32; ++kb) {
            v16bf a = *(const v16bf*)&hbuf[p][am][kb * 32 + khalf];
            acc = __builtin_amdgcn_wmma_f32_16x16x32_bf16(false, a, false, Breg[kb],
                                                          (short)0, acc, false, false);
        }

#pragma unroll
        for (int r = 0; r < 8; ++r) {
            const float v = fast_tanh(acc[r] + xp[r]);
            hpp[r][(size_t)t * H_] = v;            // fp32 hidden_states out
            hbuf[1 - p][cm + r][n] = (bf16_t)v;    // bf16 h for next step
        }
        __syncthreads();

#pragma unroll
        for (int r = 0; r < 8; ++r) xp[r] = xpn[r];
        p ^= 1;
    }
}

// ---------------------------------------------------------------------------
// Kernel 3: outputs[m,n] = sum_k hidden[m,k]*W_fc[n,k] + b_fc[n]
//   M = 131072, K = H = 256, N = O = 128.
// ---------------------------------------------------------------------------
__global__ void __launch_bounds__(256)
fc_kernel(const float* __restrict__ hid, const bf16_t* __restrict__ wfc,
          const float* __restrict__ b_fc, float* __restrict__ out) {
    const int lane  = threadIdx.x & 31;
    const int wid   = threadIdx.x >> 5;
    const int w     = blockIdx.x * 8 + wid;
    const int mtile = w >> 3;                   // 8 N-tiles
    const int ntile = w & 7;
    const int am    = mtile * 16 + (lane & 15);
    const int n     = ntile * 16 + (lane & 15);
    const int khalf = (lane >> 4) * 16;

    v8f acc = {};
#pragma unroll
    for (int kb = 0; kb < H_ / 32; ++kb) {
        const int k0 = kb * 32 + khalf;
        v16bf a;
        const float* hp = hid + (size_t)am * H_ + k0;
#pragma unroll
        for (int j = 0; j < 16; ++j) a[j] = (bf16_t)hp[j];
        v16bf b = *(const v16bf*)(wfc + (size_t)n * H_ + k0);
        acc = __builtin_amdgcn_wmma_f32_16x16x32_bf16(false, a, false, b,
                                                      (short)0, acc, false, false);
    }
    const float bias = b_fc[n];
    const int cmbase = mtile * 16 + ((lane >> 4) << 3);
#pragma unroll
    for (int r = 0; r < 8; ++r)
        out[(size_t)(cmbase + r) * O_ + n] = acc[r] + bias;
}

// ---------------------------------------------------------------------------
// Host-side launcher.  Workspace layout (bytes):
//   [0,      65536)   W_ih bf16  (256x128)
//   [65536, 196608)   W_hh bf16  (256x256)
//   [196608,262144)   W_fc bf16  (128x256)
//   [262144, +134MB)  xproj fp32 (131072x256)
// ---------------------------------------------------------------------------
extern "C" void kernel_launch(void* const* d_in, const int* in_sizes, int n_in,
                              void* d_out, int out_size, void* d_ws, size_t ws_size,
                              hipStream_t stream) {
    const float* x    = (const float*)d_in[0];
    const float* W_ih = (const float*)d_in[1];
    const float* W_hh = (const float*)d_in[2];
    const float* b_ih = (const float*)d_in[3];
    const float* b_hh = (const float*)d_in[4];
    const float* W_fc = (const float*)d_in[5];
    const float* b_fc = (const float*)d_in[6];

    float* out = (float*)d_out;                        // [B,S,O]
    float* hid = out + (size_t)B_ * S_ * O_;           // [B,S,H]

    char*   ws   = (char*)d_ws;
    bf16_t* wih  = (bf16_t*)(ws);
    bf16_t* whh  = (bf16_t*)(ws + 65536);
    bf16_t* wfc  = (bf16_t*)(ws + 196608);
    float*  xprj = (float*)(ws + 262144);

    // 0) fp32 -> bf16 weight conversion (covers max 65536 elements)
    cvt_weights_kernel<<<256, 256, 0, stream>>>(W_ih, W_hh, W_fc, wih, whh, wfc);

    // 1) input projection GEMM: 8192x16 tiles, 8 waves/block
    xproj_kernel<<<16384, 256, 0, stream>>>(x, wih, b_ih, b_hh, xprj);

    // 2) sequential scan: 4 batch groups x 16 waves
    rnn_scan_kernel<<<4, 512, 0, stream>>>(xprj, whh, hid);

    // 3) output FC GEMM: 65536 tiles, 8 waves/block
    fc_kernel<<<8192, 256, 0, stream>>>(hid, wfc, b_fc, out);
}
